// Selector_76828374990861
// MI455X (gfx1250) — matrix operand order repaired and checked by reading, not compiled, not run on hardware
//
#include <hip/hip_runtime.h>

typedef __bf16 bf16_t;
typedef __attribute__((ext_vector_type(16))) __bf16 v16bf;
typedef __attribute__((ext_vector_type(8)))  float  v8f;

#define DEV __device__ __forceinline__

namespace {

constexpr int B_    = 128;
constexpr int N_    = 200;
constexpr int K_    = 16;
constexpr int NBR   = 15;
constexpr int DNODE = 128;
constexpr int DTGT  = 128;
constexpr int PIK   = 416;                 // 386 padded to 13*32
constexpr int EROWS = B_ * N_ * NBR;       // 384000
constexpr int GROWS = B_ * N_;             // 25600

DEV bf16_t f2bf(float f) { return (bf16_t)f; }

DEV v8f vzero() {
  v8f z;
#pragma unroll
  for (int r = 0; r < 8; ++r) z[r] = 0.f;
  return z;
}

DEV v8f wmma_bf16(v16bf a, v16bf b, v8f c) {
  return __builtin_amdgcn_wmma_f32_16x16x32_bf16(false, a, false, b, (short)0, c,
                                                 false, false);
}

// A fragment (16x32 bf16) gathered from row-major f32 rows in LDS.
// lanes 0-15: M=lane, K in {0..15}; lanes 16-31: M=lane-16, K in {16..31}
DEV v16bf ld_afrag(const float* rows, int ld, int kbase) {
  const int lane = threadIdx.x & 31;
  const int m    = lane & 15;
  const int kofs = (lane & 16) ? 16 : 0;
  v16bf a;
#pragma unroll
  for (int j = 0; j < 8; ++j) {
    const int kb = 2 * j;
#pragma unroll
    for (int h = 0; h < 2; ++h)
      a[2 * j + h] = f2bf(rows[m * ld + kbase + kofs + kb + h]);
  }
  return a;
}

// A fragment with only K=0..15 valid (attention A*V, K padded to 32 with zeros)
DEV v16bf ld_afrag_pad16(const float* rows16) {
  const int lane = threadIdx.x & 31;
  const int m    = lane & 15;
  const int kofs = (lane & 16) ? 16 : 0;
  v16bf a;
#pragma unroll
  for (int j = 0; j < 8; ++j) {
    const int kb = 2 * j;
#pragma unroll
    for (int h = 0; h < 2; ++h) {
      const int kl = kofs + kb + h;
      a[2 * j + h] = (kl < 16) ? f2bf(rows16[m * 16 + kl]) : f2bf(0.f);
    }
  }
  return a;
}

// B fragment from pre-swizzled weights: one contiguous 32B load per lane.
DEV v16bf ld_bfrag(const bf16_t* w) {
  const int lane = threadIdx.x & 31;
  return *(const v16bf*)(w + (size_t)lane * 16);
}

// B fragment for attention scores: B[kk][n] = Kmat[n][col0+kk]
DEV v16bf ld_bfrag_scores(const float* kmat, int ld, int col0) {
  const int lane  = threadIdx.x & 31;
  const int n     = lane & 15;
  const int kbase = (lane & 16) ? 16 : 0;
  v16bf b;
#pragma unroll
  for (int j = 0; j < 8; ++j)
#pragma unroll
    for (int h = 0; h < 2; ++h) {
      const int kk = kbase + 2 * j + h;
      b[2 * j + h] = f2bf(kmat[n * ld + col0 + kk]);
    }
  return b;
}

// B fragment for A*V: B[kk][n] = V[kk][col0+n], rows kk>=16 are zero pad
DEV v16bf ld_bfrag_av(const float* vmat, int ld, int col0) {
  const int lane  = threadIdx.x & 31;
  const int n     = lane & 15;
  const int kbase = (lane & 16) ? 16 : 0;
  v16bf b;
#pragma unroll
  for (int j = 0; j < 8; ++j)
#pragma unroll
    for (int h = 0; h < 2; ++h) {
      const int kk = kbase + 2 * j + h;
      b[2 * j + h] = (kk < 16) ? f2bf(vmat[kk * ld + col0 + n]) : f2bf(0.f);
    }
  return b;
}

DEV void layer_norm_row(float* x, const float* g, const float* bta) {
  float m = 0.f;
  for (int c = 0; c < 256; ++c) m += x[c];
  m *= (1.f / 256.f);
  float v = 0.f;
  for (int c = 0; c < 256; ++c) { float d = x[c] - m; v += d * d; }
  v *= (1.f / 256.f);
  const float inv = rsqrtf(v + 1e-5f);
  for (int c = 0; c < 256; ++c) x[c] = (x[c] - m) * inv * g[c] + bta[c];
}

} // namespace

// ---------------------------------------------------------------------------
// Kernel 0: swizzle fp32 weights [K][N] -> bf16 B-fragment layout
//           [ktile][ntile][lane][16]
// ---------------------------------------------------------------------------
__global__ void swizzle_kernel(const float* __restrict__ W, bf16_t* __restrict__ dst,
                               int Kdim, int Ndim, int KT, int NT) {
  const int t = blockIdx.x * blockDim.x + threadIdx.x;
  if (t >= KT * NT * 32) return;
  const int lane = t & 31;
  const int nt   = (t >> 5) % NT;
  const int kt   = (t >> 5) / NT;
  bf16_t* out    = dst + (size_t)t * 16;
  const int n    = nt * 16 + (lane & 15);
  const int kb0  = kt * 32 + ((lane & 16) ? 16 : 0);
#pragma unroll
  for (int j = 0; j < 8; ++j)
#pragma unroll
    for (int h = 0; h < 2; ++h) {
      const int k = kb0 + 2 * j + h;
      out[2 * j + h] = f2bf((k < Kdim) ? W[(size_t)k * Ndim + n] : 0.f);
    }
}

// ---------------------------------------------------------------------------
// Kernel 1/2: fused feature-gather + 3-layer MLP (WMMA bf16)
//   EDGE: rows = (b,n,e), feat = [tgt | src | dst | onehot2 | pad] (416)
//   GOAL: rows = (b,n),   feat = [tgt | node] (256)
// ---------------------------------------------------------------------------
template <int KT1, bool EDGE>
__global__ void __launch_bounds__(128) mlp_kernel(
    const float* __restrict__ tgt, const float* __restrict__ nodes,
    const int* __restrict__ s_next, const int* __restrict__ edge_type,
    const int* __restrict__ node_masks,
    const bf16_t* __restrict__ W1f, const float* __restrict__ b1,
    const bf16_t* __restrict__ W2f, const float* __restrict__ b2,
    const float* __restrict__ w3, const float* __restrict__ b3,
    float* __restrict__ outv, int nrows) {
  constexpr int KIN = KT1 * 32;
  __shared__ float feat[16][PIK];
  __shared__ float h1[16][256];
  __shared__ float h2[16][128];
  const int tid  = threadIdx.x;
  const int wave = tid >> 5;
  const int lane = tid & 31;
  const int mofs = (lane & 16) ? 8 : 0;
  const int nn   = lane & 15;
  const int row0 = blockIdx.x * 16;

  // build feature tile
  for (int idx = tid; idx < 16 * KIN; idx += 128) {
    const int r = idx / KIN, c = idx % KIN;
    const int gid = row0 + r;
    float v = 0.f;
    if (gid < nrows) {
      if (EDGE) {
        const int b = gid / (N_ * NBR);
        const int rem = gid % (N_ * NBR);
        const int n = rem / NBR, e = rem % NBR;
        if (c < 128) v = tgt[b * DTGT + c];
        else if (c < 256) v = nodes[((size_t)b * N_ + n) * DNODE + (c - 128)];
        else if (c < 384) {
          const int dst = s_next[((size_t)b * N_ + n) * K_ + e];
          v = nodes[((size_t)b * N_ + dst) * DNODE + (c - 256)];
        } else if (c < 386) {
          const int et = edge_type[((size_t)b * N_ + n) * K_ + e];
          v = (c == 384) ? (et == 1 ? 1.f : 0.f) : (et == 2 ? 1.f : 0.f);
        }
      } else {
        const int b = gid / N_, n = gid % N_;
        v = (c < 128) ? tgt[b * DTGT + c]
                      : nodes[((size_t)b * N_ + n) * DNODE + (c - 128)];
      }
    }
    feat[r][c] = v;
  }
  __syncthreads();

  // layer 1: KIN -> 256, wave handles 4 column tiles
  {
    v8f acc[4];
#pragma unroll
    for (int i = 0; i < 4; ++i) acc[i] = vzero();
    for (int kt = 0; kt < KT1; ++kt) {
      v16bf a = ld_afrag(&feat[0][0], PIK, kt * 32);
#pragma unroll
      for (int i = 0; i < 4; ++i) {
        v16bf bb = ld_bfrag(W1f + ((size_t)(kt * 16 + wave * 4 + i)) * 512);
        acc[i] = wmma_bf16(a, bb, acc[i]);
      }
    }
#pragma unroll
    for (int i = 0; i < 4; ++i) {
      const int n0 = (wave * 4 + i) * 16 + nn;
      const float bb = b1[n0];
#pragma unroll
      for (int r = 0; r < 8; ++r) h1[mofs + r][n0] = fmaxf(acc[i][r] + bb, 0.f);
    }
  }
  __syncthreads();

  // layer 2: 256 -> 128, wave handles 2 column tiles
  {
    v8f acc[2];
#pragma unroll
    for (int i = 0; i < 2; ++i) acc[i] = vzero();
    for (int kt = 0; kt < 8; ++kt) {
      v16bf a = ld_afrag(&h1[0][0], 256, kt * 32);
#pragma unroll
      for (int i = 0; i < 2; ++i) {
        v16bf bb = ld_bfrag(W2f + ((size_t)(kt * 8 + wave * 2 + i)) * 512);
        acc[i] = wmma_bf16(a, bb, acc[i]);
      }
    }
#pragma unroll
    for (int i = 0; i < 2; ++i) {
      const int n0 = (wave * 2 + i) * 16 + nn;
      const float bb = b2[n0];
#pragma unroll
      for (int r = 0; r < 8; ++r) h2[mofs + r][n0] = fmaxf(acc[i][r] + bb, 0.f);
    }
  }
  __syncthreads();

  // layer 3: 128 -> 1 + masking
  if (tid < 16) {
    const int gid = row0 + tid;
    if (gid < nrows) {
      float s = b3[0];
      for (int c = 0; c < 128; ++c) s += h2[tid][c] * w3[c];
      if (EDGE) {
        const int b = gid / (N_ * NBR);
        const int rem = gid % (N_ * NBR);
        const int n = rem / NBR, e = rem % NBR;
        const int et = edge_type[((size_t)b * N_ + n) * K_ + e];
        outv[gid] = (et == 1 || et == 2) ? s : 0.f;
      } else {
        outv[gid] = (node_masks[gid] == 0) ? s : 0.f;
      }
    }
  }
}

// ---------------------------------------------------------------------------
// Kernel 3: masked log-softmax over 16 actions -> pi (d_out) and prob (ws)
// ---------------------------------------------------------------------------
__global__ void policy_softmax_kernel(const float* __restrict__ pe,
                                      const float* __restrict__ pg,
                                      const int* __restrict__ edge_type,
                                      float* __restrict__ pi_out,
                                      float* __restrict__ prob) {
  const int i = blockIdx.x * blockDim.x + threadIdx.x;
  if (i >= GROWS) return;
  float x[16];
#pragma unroll
  for (int k = 0; k < 15; ++k) x[k] = pe[(size_t)i * 15 + k];
  x[15] = pg[i];
  const int* et = edge_type + (size_t)i * 16;
  float mx = -3.0e38f;
  int nv = 0;
#pragma unroll
  for (int k = 0; k < 16; ++k)
    if (et[k] != 0) { ++nv; mx = fmaxf(mx, x[k]); }
  float* po = pi_out + (size_t)i * 16;
  float* pr = prob + (size_t)i * 16;
  if (nv == 0) {
#pragma unroll
    for (int k = 0; k < 16; ++k) { po[k] = -1.0e30f; pr[k] = 0.f; }
    return;
  }
  float e[16], sum = 0.f;
#pragma unroll
  for (int k = 0; k < 16; ++k) {
    e[k] = (et[k] != 0) ? __expf(x[k] - mx) : 0.f;
    sum += e[k];
  }
  const float lse = __logf(sum), inv = 1.f / sum;
#pragma unroll
  for (int k = 0; k < 16; ++k) {
    po[k] = (et[k] != 0) ? (x[k] - mx - lse) : -1.0e30f;
    pr[k] = e[k] * inv;
  }
}

// ---------------------------------------------------------------------------
// Kernel 4: probability propagation scan (HORIZON-1 steps, LDS scatter-add)
// ---------------------------------------------------------------------------
__global__ void __launch_bounds__(256) sample_kernel(
    const float* __restrict__ prob, const int* __restrict__ s_next,
    const float* __restrict__ initn, float* __restrict__ nprob,
    int* __restrict__ nseq) {
  __shared__ float prev[2 * N_], nxt[2 * N_], acc[2 * N_], mxp[2 * N_];
  __shared__ int seq[2 * N_];
  const int b = blockIdx.x, tid = threadIdx.x;
  for (int n = tid; n < 2 * N_; n += 256) {
    const float p = (n < N_) ? ((initn[(size_t)b * N_ + n] != 0.f) ? 1.f : 0.f) : 0.f;
    prev[n] = p; acc[n] = p; mxp[n] = p; seq[n] = (p != 0.f) ? 1 : 0; nxt[n] = 0.f;
  }
  __syncthreads();
  for (int step = 1; step < 15; ++step) {
    for (int n = tid; n < 2 * N_; n += 256) {
      const float p = prev[n];
      if (p != 0.f) {
        if (n < N_) {
          const int* sn = s_next + ((size_t)b * N_ + n) * 16;
          const float* pr = prob + ((size_t)b * N_ + n) * 16;
#pragma unroll
          for (int k = 0; k < 16; ++k) {
            const float t = p * pr[k];
            if (t != 0.f) atomicAdd(&nxt[sn[k]], t);
          }
        } else {
          atomicAdd(&nxt[n], p);   // dummy self-loop
        }
      }
    }
    __syncthreads();
    for (int n = tid; n < 2 * N_; n += 256) {
      const float v = nxt[n];
      acc[n] += v;
      if (v != 0.f && mxp[n] < v) seq[n] = step + 1;
      mxp[n] = fmaxf(mxp[n], v);
      prev[n] = v;
      nxt[n] = 0.f;
    }
    __syncthreads();
  }
  for (int n = tid; n < N_; n += 256) {
    nprob[(size_t)b * N_ + n] = acc[n];
    nseq[(size_t)b * N_ + n] = seq[n];
  }
}

// ---------------------------------------------------------------------------
// Kernel 5: top-16 selection + noi_mask + transformer input build
// ---------------------------------------------------------------------------
__global__ void __launch_bounds__(32) topk_kernel(
    const float* __restrict__ nprob, const int* __restrict__ nseq,
    const float* __restrict__ nodes, const float* __restrict__ tgt,
    const float* __restrict__ posE, int* __restrict__ noim,
    float* __restrict__ maskO, float* __restrict__ xtf) {
  __shared__ float vals[N_];
  __shared__ int sel[16];
  const int b = blockIdx.x, lane = threadIdx.x;
  for (int n = lane; n < N_; n += 32) vals[n] = nprob[(size_t)b * N_ + n];
  __syncthreads();
  for (int j = 0; j < 16; ++j) {
    float bv = -2.f;
    int bi = 0;
    for (int n = lane; n < N_; n += 32) {
      const float v = vals[n];
      if (v > bv || (v == bv && n < bi)) { bv = v; bi = n; }
    }
    for (int o = 16; o; o >>= 1) {
      const float ov = __shfl_xor(bv, o, 32);
      const int oi = __shfl_xor(bi, o, 32);
      if (ov > bv || (ov == bv && oi < bi)) { bv = ov; bi = oi; }
    }
    if (lane == 0) { sel[j] = bi; vals[bi] = -1.f; }
    __syncthreads();
  }
  if (lane < 16) {
    const int idx = sel[lane];
    const int sq = nseq[(size_t)b * N_ + idx];
    noim[b * 16 + lane] = (sq != 0) ? 1 : 0;
    maskO[b * 16 + lane] = (sq != 0) ? 1.f : 0.f;
  }
  for (int t = lane; t < 16 * 256; t += 32) {
    const int j = t >> 8, c = t & 255;
    const int idx = sel[j];
    int sq = nseq[(size_t)b * N_ + idx];
    sq = sq < 0 ? 0 : (sq > 16 ? 16 : sq);
    float v;
    if (c < 128) v = nodes[((size_t)b * N_ + idx) * DNODE + c] + posE[sq * 128 + c];
    else v = tgt[(size_t)b * DTGT + (c - 128)];
    xtf[(((size_t)b * 16) + j) * 256 + c] = v;
  }
}

// ---------------------------------------------------------------------------
// Kernel 6: full 2-layer transformer encoder + masked mean, one block / sample
// ---------------------------------------------------------------------------
struct TfLayer {
  const bf16_t *Wq, *Wk, *Wv, *Wo, *Wf1, *Wf2;
  const float *bq, *bk, *bv, *bo, *bf1, *bf2, *g1, *b1, *g2, *b2;
};
struct TfArgs {
  const float* xtf;
  const int* noim;
  const bf16_t* Wproj;
  const float* bproj;
  TfLayer L[2];
  float* agg;
};

__global__ void __launch_bounds__(128) transformer_kernel(TfArgs A) {
  extern __shared__ float smem[];
  float* xb  = smem;          // 16x256
  float* res = xb + 4096;     // 16x256
  float* qb  = res + 4096;    // 16x256
  float* kb  = qb + 4096;     // 16x256
  float* vb  = kb + 4096;     // 16x256
  float* ob  = vb + 4096;     // 16x256
  float* hb  = ob + 4096;     // 16x1024
  float* sb  = hb + 16384;    // 4 waves x 16x16 attn weights
  float* msk = sb + 1024;     // 16

  const int b    = blockIdx.x;
  const int tid  = threadIdx.x;
  const int wave = tid >> 5;
  const int lane = tid & 31;
  const int mofs = (lane & 16) ? 8 : 0;
  const int nn   = lane & 15;

  for (int i = tid; i < 4096; i += 128) res[i] = A.xtf[(size_t)b * 4096 + i];
  if (tid < 16) msk[tid] = (A.noim[b * 16 + tid] != 0) ? 1.f : 0.f;
  __syncthreads();

  // input projection: xb = res @ Wproj + bproj
  {
    v8f acc[4];
#pragma unroll
    for (int i = 0; i < 4; ++i) acc[i] = vzero();
    for (int kt = 0; kt < 8; ++kt) {
      v16bf a = ld_afrag(res, 256, kt * 32);
#pragma unroll
      for (int i = 0; i < 4; ++i)
        acc[i] = wmma_bf16(a, ld_bfrag(A.Wproj + ((size_t)(kt * 16 + wave * 4 + i)) * 512), acc[i]);
    }
#pragma unroll
    for (int i = 0; i < 4; ++i) {
      const int n0 = (wave * 4 + i) * 16 + nn;
      const float bv_ = A.bproj[n0];
#pragma unroll
      for (int r = 0; r < 8; ++r) xb[(mofs + r) * 256 + n0] = acc[i][r] + bv_;
    }
  }
  __syncthreads();

  for (int li = 0; li < 2; ++li) {
    const TfLayer& Lp = A.L[li];
    for (int i = tid; i < 4096; i += 128) res[i] = xb[i];   // residual
    __syncthreads();

    // Q, K, V projections
    {
      v8f aq[4], ak[4], av[4];
#pragma unroll
      for (int i = 0; i < 4; ++i) { aq[i] = vzero(); ak[i] = vzero(); av[i] = vzero(); }
      for (int kt = 0; kt < 8; ++kt) {
        v16bf a = ld_afrag(xb, 256, kt * 32);
#pragma unroll
        for (int i = 0; i < 4; ++i) {
          const size_t fo = ((size_t)(kt * 16 + wave * 4 + i)) * 512;
          aq[i] = wmma_bf16(a, ld_bfrag(Lp.Wq + fo), aq[i]);
          ak[i] = wmma_bf16(a, ld_bfrag(Lp.Wk + fo), ak[i]);
          av[i] = wmma_bf16(a, ld_bfrag(Lp.Wv + fo), av[i]);
        }
      }
#pragma unroll
      for (int i = 0; i < 4; ++i) {
        const int n0 = (wave * 4 + i) * 16 + nn;
        const float q0 = Lp.bq[n0], k0 = Lp.bk[n0], v0 = Lp.bv[n0];
#pragma unroll
        for (int r = 0; r < 8; ++r) {
          qb[(mofs + r) * 256 + n0] = aq[i][r] + q0;
          kb[(mofs + r) * 256 + n0] = ak[i][r] + k0;
          vb[(mofs + r) * 256 + n0] = av[i][r] + v0;
        }
      }
    }
    __syncthreads();

    // attention: wave handles heads {2w, 2w+1}; hd = 32 = one k-tile
    for (int hi = 0; hi < 2; ++hi) {
      const int hh = wave * 2 + hi;
      v16bf aq = ld_afrag(qb, 256, hh * 32);
      v16bf bk_ = ld_bfrag_scores(kb, 256, hh * 32);
      v8f s = wmma_bf16(aq, bk_, vzero());
      const float mv = msk[nn];
      float aw[8];
#pragma unroll
      for (int r = 0; r < 8; ++r) {
        const float sv = s[r] * 0.17677669529663687f;   // 1/sqrt(32)
        aw[r] = (mv != 0.f) ? sv : -1.0e9f;
      }
#pragma unroll
      for (int r = 0; r < 8; ++r) {  // softmax over 16 key lanes
        float m2 = aw[r];
        for (int o = 8; o; o >>= 1) m2 = fmaxf(m2, __shfl_xor(m2, o, 32));
        const float e2 = __expf(aw[r] - m2);
        float s2 = e2;
        for (int o = 8; o; o >>= 1) s2 += __shfl_xor(s2, o, 32);
        aw[r] = e2 / s2;
      }
      float* sw = sb + wave * 256;
#pragma unroll
      for (int r = 0; r < 8; ++r) sw[(mofs + r) * 16 + nn] = aw[r];
      __syncthreads();
      v16bf aa = ld_afrag_pad16(sw);
#pragma unroll
      for (int t2 = 0; t2 < 2; ++t2) {
        v16bf bv_ = ld_bfrag_av(vb, 256, hh * 32 + t2 * 16);
        v8f ov = wmma_bf16(aa, bv_, vzero());
        const int n0 = hh * 32 + t2 * 16 + nn;
#pragma unroll
        for (int r = 0; r < 8; ++r) ob[(mofs + r) * 256 + n0] = ov[r];
      }
    }
    __syncthreads();

    // attn_out @ Wo + bo + residual -> xb, then LN1
    {
      v8f acc[4];
#pragma unroll
      for (int i = 0; i < 4; ++i) acc[i] = vzero();
      for (int kt = 0; kt < 8; ++kt) {
        v16bf a = ld_afrag(ob, 256, kt * 32);
#pragma unroll
        for (int i = 0; i < 4; ++i)
          acc[i] = wmma_bf16(a, ld_bfrag(Lp.Wo + ((size_t)(kt * 16 + wave * 4 + i)) * 512), acc[i]);
      }
#pragma unroll
      for (int i = 0; i < 4; ++i) {
        const int n0 = (wave * 4 + i) * 16 + nn;
        const float bo_ = Lp.bo[n0];
#pragma unroll
        for (int r = 0; r < 8; ++r) {
          const int idx = (mofs + r) * 256 + n0;
          xb[idx] = res[idx] + acc[i][r] + bo_;
        }
      }
    }
    __syncthreads();
    if (tid < 16) layer_norm_row(xb + tid * 256, Lp.g1, Lp.b1);
    __syncthreads();
    for (int i = tid; i < 4096; i += 128) res[i] = xb[i];
    __syncthreads();

    // FFN1: hb = relu(xb @ Wf1 + bf1), 16 n-tiles per wave in 4 chunks
    for (int ch = 0; ch < 4; ++ch) {
      v8f acc[4];
#pragma unroll
      for (int i = 0; i < 4; ++i) acc[i] = vzero();
      for (int kt = 0; kt < 8; ++kt) {
        v16bf a = ld_afrag(xb, 256, kt * 32);
#pragma unroll
        for (int i = 0; i < 4; ++i) {
          const int nt = wave * 16 + ch * 4 + i;
          acc[i] = wmma_bf16(a, ld_bfrag(Lp.Wf1 + ((size_t)(kt * 64 + nt)) * 512), acc[i]);
        }
      }
#pragma unroll
      for (int i = 0; i < 4; ++i) {
        const int n0 = (wave * 16 + ch * 4 + i) * 16 + nn;
        const float bf_ = Lp.bf1[n0];
#pragma unroll
        for (int r = 0; r < 8; ++r) hb[(mofs + r) * 1024 + n0] = fmaxf(acc[i][r] + bf_, 0.f);
      }
    }
    __syncthreads();

    // FFN2: xb = res + hb @ Wf2 + bf2, then LN2
    {
      v8f acc[4];
#pragma unroll
      for (int i = 0; i < 4; ++i) acc[i] = vzero();
      for (int kt = 0; kt < 32; ++kt) {
        v16bf a = ld_afrag(hb, 1024, kt * 32);
#pragma unroll
        for (int i = 0; i < 4; ++i)
          acc[i] = wmma_bf16(a, ld_bfrag(Lp.Wf2 + ((size_t)(kt * 16 + wave * 4 + i)) * 512), acc[i]);
      }
#pragma unroll
      for (int i = 0; i < 4; ++i) {
        const int n0 = (wave * 4 + i) * 16 + nn;
        const float bf_ = Lp.bf2[n0];
#pragma unroll
        for (int r = 0; r < 8; ++r) {
          const int idx = (mofs + r) * 256 + n0;
          xb[idx] = res[idx] + acc[i][r] + bf_;
        }
      }
    }
    __syncthreads();
    if (tid < 16) layer_norm_row(xb + tid * 256, Lp.g2, Lp.b2);
    __syncthreads();
  }

  // masked mean over T=16
  float denom = 0.f;
#pragma unroll
  for (int t = 0; t < 16; ++t) denom += msk[t];
  denom = fmaxf(denom, 1.f);
  for (int c = tid; c < 256; c += 128) {
    float s = 0.f;
    for (int t = 0; t < 16; ++t) s += msk[t] * xb[t * 256 + c];
    A.agg[(size_t)b * 256 + c] = s / denom;
  }
}

// ---------------------------------------------------------------------------
extern "C" void kernel_launch(void* const* d_in, const int* in_sizes, int n_in,
                              void* d_out, int out_size, void* d_ws, size_t ws_size,
                              hipStream_t stream) {
  (void)in_sizes; (void)n_in; (void)out_size; (void)ws_size;
  const float* tgt   = (const float*)d_in[0];
  const float* nodes = (const float*)d_in[1];
  const int*   nmask = (const int*)d_in[2];
  const int*   snext = (const int*)d_in[3];
  const int*   etype = (const int*)d_in[4];
  const float* initn = (const float*)d_in[5];
  const float *piW1 = (const float*)d_in[6],  *piB1 = (const float*)d_in[7];
  const float *piW2 = (const float*)d_in[8],  *piB2 = (const float*)d_in[9];
  const float *piW3 = (const float*)d_in[10], *piB3 = (const float*)d_in[11];
  const float *gW1  = (const float*)d_in[12], *gB1  = (const float*)d_in[13];
  const float *gW2  = (const float*)d_in[14], *gB2  = (const float*)d_in[15];
  const float *gW3  = (const float*)d_in[16], *gB3  = (const float*)d_in[17];
  const float* posE = (const float*)d_in[18];
  // per-layer leaves, jax alpha order: Wf1,Wf2,Wk,Wo,Wq,Wv,b1,b2,bf1,bf2,bk,bo,bq,bv,g1,g2
  const float* LW[2][16];
  for (int l = 0; l < 2; ++l)
    for (int j = 0; j < 16; ++j) LW[l][j] = (const float*)d_in[19 + l * 16 + j];
  const float *prW = (const float*)d_in[51], *prB = (const float*)d_in[52];

  char* ws = (char*)d_ws;
  size_t off = 0;
  auto alloc = [&](size_t bytes) -> char* {
    char* p = ws + off;
    off = (off + bytes + 255) & ~(size_t)255;
    return p;
  };

  bf16_t* W1pf = (bf16_t*)alloc((size_t)13 * 16 * 512 * 2);
  bf16_t* W2pf = (bf16_t*)alloc((size_t)8 * 8 * 512 * 2);
  bf16_t* G1f  = (bf16_t*)alloc((size_t)8 * 16 * 512 * 2);
  bf16_t* G2f  = (bf16_t*)alloc((size_t)8 * 8 * 512 * 2);
  bf16_t* PRf  = (bf16_t*)alloc((size_t)8 * 16 * 512 * 2);
  bf16_t* Lf[2][6];  // Wq, Wk, Wv, Wo, Wf1, Wf2
  for (int l = 0; l < 2; ++l) {
    for (int j = 0; j < 4; ++j) Lf[l][j] = (bf16_t*)alloc((size_t)8 * 16 * 512 * 2);
    Lf[l][4] = (bf16_t*)alloc((size_t)8 * 64 * 512 * 2);
    Lf[l][5] = (bf16_t*)alloc((size_t)32 * 16 * 512 * 2);
  }
  float* pe    = (float*)alloc((size_t)EROWS * 4);
  float* pg    = (float*)alloc((size_t)GROWS * 4);
  float* prob  = (float*)alloc((size_t)GROWS * 16 * 4);
  float* nprob = (float*)alloc((size_t)GROWS * 4);
  int*   nseq  = (int*)alloc((size_t)GROWS * 4);
  int*   noim  = (int*)alloc((size_t)B_ * 16 * 4);
  float* xtf   = (float*)alloc((size_t)B_ * 16 * 256 * 4);

  auto swz = [&](const float* W, bf16_t* dst, int Kd, int Nd, int KT, int NT) {
    const int total = KT * NT * 32;
    swizzle_kernel<<<(total + 255) / 256, 256, 0, stream>>>(W, dst, Kd, Nd, KT, NT);
  };
  swz(piW1, W1pf, 386, 256, 13, 16);
  swz(piW2, W2pf, 256, 128, 8, 8);
  swz(gW1, G1f, 256, 256, 8, 16);
  swz(gW2, G2f, 256, 128, 8, 8);
  swz(prW, PRf, 256, 256, 8, 16);
  for (int l = 0; l < 2; ++l) {
    swz(LW[l][4], Lf[l][0], 256, 256, 8, 16);   // Wq
    swz(LW[l][2], Lf[l][1], 256, 256, 8, 16);   // Wk
    swz(LW[l][5], Lf[l][2], 256, 256, 8, 16);   // Wv
    swz(LW[l][3], Lf[l][3], 256, 256, 8, 16);   // Wo
    swz(LW[l][0], Lf[l][4], 256, 1024, 8, 64);  // Wf1
    swz(LW[l][1], Lf[l][5], 1024, 256, 32, 16); // Wf2
  }

  mlp_kernel<13, true><<<EROWS / 16, 128, 0, stream>>>(
      tgt, nodes, snext, etype, nmask, W1pf, piB1, W2pf, piB2, piW3, piB3, pe, EROWS);
  mlp_kernel<8, false><<<GROWS / 16, 128, 0, stream>>>(
      tgt, nodes, snext, etype, nmask, G1f, gB1, G2f, gB2, gW3, gB3, pg, GROWS);

  float* outF  = (float*)d_out;
  float* aggO  = outF;                         // (128, 256)
  float* maskO = outF + 128 * 256;             // (128, 16)
  float* piO   = outF + 128 * 256 + 128 * 16;  // (128, 200, 16)

  policy_softmax_kernel<<<GROWS / 256, 256, 0, stream>>>(pe, pg, etype, piO, prob);
  sample_kernel<<<B_, 256, 0, stream>>>(prob, snext, initn, nprob, nseq);
  topk_kernel<<<B_, 32, 0, stream>>>(nprob, nseq, nodes, tgt, posE, noim, maskO, xtf);

  TfArgs TA;
  TA.xtf = xtf; TA.noim = noim; TA.Wproj = PRf; TA.bproj = prB; TA.agg = aggO;
  for (int l = 0; l < 2; ++l) {
    TA.L[l].Wq = Lf[l][0]; TA.L[l].Wk = Lf[l][1]; TA.L[l].Wv = Lf[l][2];
    TA.L[l].Wo = Lf[l][3]; TA.L[l].Wf1 = Lf[l][4]; TA.L[l].Wf2 = Lf[l][5];
    TA.L[l].bq = LW[l][12]; TA.L[l].bk = LW[l][10]; TA.L[l].bv = LW[l][13];
    TA.L[l].bo = LW[l][11]; TA.L[l].bf1 = LW[l][8]; TA.L[l].bf2 = LW[l][9];
    TA.L[l].g1 = LW[l][14]; TA.L[l].b1 = LW[l][6];
    TA.L[l].g2 = LW[l][15]; TA.L[l].b2 = LW[l][7];
  }
  // dynamic LDS: 6x(16x256) + 16x1024 + 4x256 + 16 floats = 168 KB (of 320 KB WGP LDS)
  transformer_kernel<<<B_, 128, 42000 * sizeof(float), stream>>>(TA);
}
